// RPN_76072460746888
// MI455X (gfx1250) — matrix operand order repaired and checked by reading, not compiled, not run on hardware
//
#include <hip/hip_runtime.h>

#define PT 80997        // total pixels across levels (one batch)
#define AT 242991       // total anchors (3*PT)
#define SELN 4741       // selected after per-level top-k: 1000*4 + 741
#define NPS 349184      // sum of per-level pow2-padded sort sizes
#define NPADN 8192      // padded NMS sort size
#define MASKW 152       // ceil(4741/32)=149, padded to 152
#define NEGINF (-__builtin_inff())
#define BBOX_CLIP_F 4.135166556742356f

typedef __bf16 v16bf __attribute__((ext_vector_type(16)));
typedef float  v8f   __attribute__((ext_vector_type(8)));
typedef __attribute__((__vector_size__(4 * sizeof(int)))) int vi4;

#if defined(__has_builtin)
#if __has_builtin(__builtin_amdgcn_global_load_async_to_lds_b128)
#define HAVE_ASYNC_LDS 1
#endif
#endif
#ifndef HAVE_ASYNC_LDS
#define HAVE_ASYNC_LDS 0
#endif

__device__ __forceinline__ unsigned short f2bf(float x) {
  unsigned u = __float_as_uint(x);
  unsigned r = u + 0x7fffu + ((u >> 16) & 1u);
  return (unsigned short)(r >> 16);
}

#if HAVE_ASYNC_LDS
// copy 32 contiguous bytes global -> LDS through the async engine (ASYNCcnt)
__device__ __forceinline__ void async_cp32(const void* g, void* l) {
  __attribute__((address_space(1))) vi4* gp =
      (__attribute__((address_space(1))) vi4*)(unsigned long long)g;
  __attribute__((address_space(3))) vi4* lp =
      (__attribute__((address_space(3))) vi4*)(unsigned int)(unsigned long long)l;
  __builtin_amdgcn_global_load_async_to_lds_b128(gp, lp, 0, 0);
  __builtin_amdgcn_global_load_async_to_lds_b128(gp, lp, 16, 0);
}
__device__ __forceinline__ void async_wait0() {
#if __has_builtin(__builtin_amdgcn_s_wait_asynccnt)
  __builtin_amdgcn_s_wait_asynccnt(0);
#else
  asm volatile("s_wait_asynccnt 0x0" ::: "memory");
#endif
}
#endif

// ---------------- input f32 NCHW -> bf16 pixel-major channel-last ----------------
__global__ void k_convert_input(const float* __restrict__ f, unsigned short* __restrict__ fbf,
                                int HW, int Poff) {
  int idx = blockIdx.x * 256 + threadIdx.x;
  if (idx >= 2 * 32 * HW) return;
  int b  = idx / (32 * HW);
  int r  = idx % (32 * HW);
  int c8 = r / HW;
  int p  = r % HW;
  const float* src = f + ((size_t)(b * 256 + c8 * 8)) * HW + p;
  unsigned short o[8];
#pragma unroll
  for (int i = 0; i < 8; ++i) o[i] = f2bf(src[(size_t)i * HW]);
  uint4 st;
  st.x = (unsigned)o[0] | ((unsigned)o[1] << 16);
  st.y = (unsigned)o[2] | ((unsigned)o[3] << 16);
  st.z = (unsigned)o[4] | ((unsigned)o[5] << 16);
  st.w = (unsigned)o[6] | ((unsigned)o[7] << 16);
  *(uint4*)(fbf + ((size_t)b * PT + Poff + p) * 256 + c8 * 8) = st;
}

// ------------- conv weights f32 [cout,cin,3,3] -> bf16 WMMA A-fragment order -------------
__global__ void k_convert_weights(const float* __restrict__ w, unsigned short* __restrict__ wfrag) {
  int idx = blockIdx.x * 256 + threadIdx.x;
  if (idx >= 16 * 72 * 32 * 16) return;
  int e    = idx & 15;
  int lane = (idx >> 4) & 31;
  int t2   = idx >> 9;
  int kidx = t2 % 72;
  int cb   = t2 / 72;
  int tap = kidx >> 3, cs = kidx & 7;
  int K = (lane >> 4) * 16 + e;
  int cin  = cs * 32 + K;
  int cout = cb * 16 + (lane & 15);
  int ky = tap / 3, kx = tap % 3;
  wfrag[idx] = f2bf(w[((size_t)(cout * 256 + cin)) * 9 + ky * 3 + kx]);
}

// ---------------- 3x3 conv + bias + ReLU, implicit GEMM via WMMA bf16 ----------------
// Wave tile: 16 couts x 64 pixels (4 independent accumulators, A reused 4x per k-step).
// B tiles staged in LDS via async global->LDS copies when available.
__global__ void __launch_bounds__(256) k_conv3x3(
    const unsigned short* __restrict__ fbf, const unsigned short* __restrict__ wfrag,
    const float* __restrict__ bias, unsigned short* __restrict__ tbf,
    int H, int W, int Poff) {
  __shared__ __align__(16) unsigned short ldsB[4 * 8 * 32 * 16];  // 4 pixel-tiles x 8KB
  const int HW   = H * W;
  const int tid  = threadIdx.x;
  const int lane = tid & 31;
  const int wave = tid >> 5;
  const int b    = blockIdx.z;
  const int pix0 = blockIdx.x * 64;
  const int coutBase = blockIdx.y * 128 + wave * 16;
  const int coutblk  = coutBase >> 4;

  const unsigned short* fb = fbf + ((size_t)b * PT + Poff) * 256;

  // LDS fill decomposition: one slot per (thread, tile); slot = cs*32 + l within tile
  const int fl    = tid & 31;
  const int fn    = fl & 15;
  const int fcin0 = (tid >> 5) * 32 + ((fl >> 4) << 4);  // lanes>=16 hold K=16..31
  int fp[4], fy[4], fx[4];
#pragma unroll
  for (int t = 0; t < 4; ++t) {
    fp[t] = pix0 + t * 16 + fn;
    fy[t] = (fp[t] < HW) ? (fp[t] / W) : 0;
    fx[t] = (fp[t] < HW) ? (fp[t] % W) : 0;
  }

  // lane-fixed base pointers; all k-step deltas become constant instruction offsets
  const uint4* wbase = (const uint4*)wfrag + ((size_t)(coutblk * 72) * 32 + lane) * 2;
  const uint4* lbase = (const uint4*)ldsB + lane * 2;
  uint4*       fbase = (uint4*)ldsB + tid * 2;

  v8f acc[4];
#pragma unroll
  for (int t = 0; t < 4; ++t)
#pragma unroll
    for (int i = 0; i < 8; ++i) acc[t][i] = 0.0f;

#pragma unroll
  for (int tap = 0; tap < 9; ++tap) {
    const int ky = tap / 3 - 1, kx = tap % 3 - 1;
#pragma unroll
    for (int t = 0; t < 4; ++t) {
      uint4* d = fbase + t * 512;   // tile stride = 8KB = 512 uint4
      bool inb = false;
      const uint4* src = (const uint4*)fb;
      if (fp[t] < HW) {
        int y = fy[t] + ky, x = fx[t] + kx;
        if (y >= 0 && y < H && x >= 0 && x < W) {
          inb = true;
          src = (const uint4*)(fb + ((size_t)(y * W + x)) * 256 + fcin0);
        }
      }
#if HAVE_ASYNC_LDS
      if (inb) {
        async_cp32(src, d);
      } else {
        uint4 z = make_uint4(0, 0, 0, 0);
        d[0] = z; d[1] = z;
      }
#else
      uint4 lo = make_uint4(0, 0, 0, 0), hi = make_uint4(0, 0, 0, 0);
      if (inb) { lo = src[0]; hi = src[1]; }
      d[0] = lo; d[1] = hi;
#endif
    }
#if HAVE_ASYNC_LDS
    async_wait0();
#endif
    __syncthreads();
#pragma unroll
    for (int cs = 0; cs < 8; ++cs) {
      union { v16bf v; uint4 u[2]; } A, B0, B1, B2, B3;
      const uint4* wa = wbase + (tap * 8 + cs) * 64;  // k-step stride = 1KB = 64 uint4
      A.u[0] = wa[0]; A.u[1] = wa[1];
      const uint4* l0 = lbase + cs * 64;
      B0.u[0] = l0[0];    B0.u[1] = l0[1];
      B1.u[0] = l0[512];  B1.u[1] = l0[513];
      B2.u[0] = l0[1024]; B2.u[1] = l0[1025];
      B3.u[0] = l0[1536]; B3.u[1] = l0[1537];
      acc[0] = __builtin_amdgcn_wmma_f32_16x16x32_bf16(false, A.v, false, B0.v,
                                                       (short)0, acc[0], false, false);
      acc[1] = __builtin_amdgcn_wmma_f32_16x16x32_bf16(false, A.v, false, B1.v,
                                                       (short)0, acc[1], false, false);
      acc[2] = __builtin_amdgcn_wmma_f32_16x16x32_bf16(false, A.v, false, B2.v,
                                                       (short)0, acc[2], false, false);
      acc[3] = __builtin_amdgcn_wmma_f32_16x16x32_bf16(false, A.v, false, B3.v,
                                                       (short)0, acc[3], false, false);
    }
    __syncthreads();
  }

  // epilogue: C/D layout -> per lane 8 consecutive channels of one pixel per tile
  const int n  = lane & 15;
  const int hh = lane >> 4;
  float bs[8];
#pragma unroll
  for (int r = 0; r < 8; ++r) bs[r] = bias[coutBase + r + 8 * hh];
#pragma unroll
  for (int t = 0; t < 4; ++t) {
    const int p = pix0 + t * 16 + n;
    if (p < HW) {
      unsigned short o[8];
#pragma unroll
      for (int r = 0; r < 8; ++r) {
        float v = acc[t][r] + bs[r];
        v = v > 0.0f ? v : 0.0f;
        o[r] = f2bf(v);
      }
      uint4 st;
      st.x = (unsigned)o[0] | ((unsigned)o[1] << 16);
      st.y = (unsigned)o[2] | ((unsigned)o[3] << 16);
      st.z = (unsigned)o[4] | ((unsigned)o[5] << 16);
      st.w = (unsigned)o[6] | ((unsigned)o[7] << 16);
      *(uint4*)(tbf + ((size_t)b * PT + Poff + p) * 256 + coutBase + hh * 8) = st;
    }
  }
}

// ---------------- 1x1 heads + anchor decode + clip + validity ----------------
__global__ void __launch_bounds__(256) k_heads(
    const unsigned short* __restrict__ tbf,
    const float* __restrict__ clsw, const float* __restrict__ clsb,
    const float* __restrict__ boxw, const float* __restrict__ boxb,
    float* __restrict__ scores, float* __restrict__ boxes, unsigned char* __restrict__ valid,
    int H, int W, int Poff, int Aoff, float strideL, float asize) {
  __shared__ float wsh[15 * 256 + 16];
  const int tid = threadIdx.x;
  for (int i = tid; i < 15 * 256; i += 256) {
    int row = i >> 8, c = i & 255;
    wsh[i] = (row < 3) ? clsw[row * 256 + c] : boxw[(row - 3) * 256 + c];
  }
  if (tid < 15) wsh[3840 + tid] = (tid < 3) ? clsb[tid] : boxb[tid - 3];
  __syncthreads();

  const int HW = H * W;
  const int idx = blockIdx.x * 256 + tid;
  if (idx >= 2 * HW) return;
  const int b = idx / HW, p = idx % HW;
  const uint4* t = (const uint4*)(tbf + ((size_t)b * PT + Poff + p) * 256);

  float accum[15];
#pragma unroll
  for (int k = 0; k < 15; ++k) accum[k] = 0.0f;
  for (int c8 = 0; c8 < 32; ++c8) {
    uint4 u = t[c8];
    float f[8];
    f[0] = __uint_as_float(u.x << 16); f[1] = __uint_as_float(u.x & 0xffff0000u);
    f[2] = __uint_as_float(u.y << 16); f[3] = __uint_as_float(u.y & 0xffff0000u);
    f[4] = __uint_as_float(u.z << 16); f[5] = __uint_as_float(u.z & 0xffff0000u);
    f[6] = __uint_as_float(u.w << 16); f[7] = __uint_as_float(u.w & 0xffff0000u);
#pragma unroll
    for (int k = 0; k < 15; ++k) {
      const float* wr = &wsh[k * 256 + c8 * 8];
      float s = 0.0f;
#pragma unroll
      for (int i = 0; i < 8; ++i) s += f[i] * wr[i];
      accum[k] += s;
    }
  }

  const float x = (float)(p % W), y = (float)(p / W);
  const float hr[3] = {0.70710678118654752f, 1.0f, 1.41421356237309505f};
  for (int a = 0; a < 3; ++a) {
    float score = accum[a] + wsh[3840 + a];
    float dx = accum[3 + a * 4 + 0] + wsh[3843 + a * 4 + 0];
    float dy = accum[3 + a * 4 + 1] + wsh[3843 + a * 4 + 1];
    float dw = accum[3 + a * 4 + 2] + wsh[3843 + a * 4 + 2];
    float dh = accum[3 + a * 4 + 3] + wsh[3843 + a * 4 + 3];
    dw = fminf(dw, BBOX_CLIP_F); dh = fminf(dh, BBOX_CLIP_F);
    float wa = asize / hr[a], ha = asize * hr[a];
    float cx = dx * wa + x * strideL, cy = dy * ha + y * strideL;
    float ww = expf(dw) * wa, hh = expf(dh) * ha;
    float x0 = fminf(fmaxf(cx - 0.5f * ww, 0.0f), 1216.0f);
    float y0 = fminf(fmaxf(cy - 0.5f * hh, 0.0f), 800.0f);
    float x1 = fminf(fmaxf(cx + 0.5f * ww, 0.0f), 1216.0f);
    float y1 = fminf(fmaxf(cy + 0.5f * hh, 0.0f), 800.0f);
    size_t gi = (size_t)b * AT + Aoff + p * 3 + a;
    scores[gi] = score;
    float* bx = boxes + gi * 4;
    bx[0] = x0; bx[1] = y0; bx[2] = x1; bx[3] = y1;
    valid[gi] = (((x1 - x0) >= 1e-3f) && ((y1 - y0) >= 1e-3f)) ? 1 : 0;
  }
}

// ---------------- per-level sort buffer init ----------------
__global__ void k_sortinit(const float* __restrict__ scores, float* __restrict__ key,
                           int* __restrict__ val, int nA, int Aoff, int npad, int npadoff) {
  int idx = blockIdx.x * 256 + threadIdx.x;
  if (idx >= 2 * npad) return;
  int b = idx / npad, i = idx % npad;
  float kk = (i < nA) ? scores[(size_t)b * AT + Aoff + i] : NEGINF;
  key[(size_t)b * NPS + npadoff + i] = kk;
  val[(size_t)b * NPS + npadoff + i] = i;
}

// ---------------- bitonic pass (descending) ----------------
__global__ void k_bitonic(float* __restrict__ key, int* __restrict__ val,
                          int strideB, int n, int k, int j) {
  int idx = blockIdx.x * 256 + threadIdx.x;
  if (idx >= 2 * n) return;
  int b = idx / n, i = idx % n;
  int l = i ^ j;
  if (l <= i) return;
  size_t base = (size_t)b * strideB;
  float ki = key[base + i], kl = key[base + l];
  bool up = ((i & k) == 0);
  bool sw = up ? (ki < kl) : (ki > kl);
  if (sw) {
    key[base + i] = kl; key[base + l] = ki;
    int vi = val[base + i]; val[base + i] = val[base + l]; val[base + l] = vi;
  }
}

// ---------------- gather per-level top-k into combined selected set ----------------
__global__ void k_gather(const float* __restrict__ scores, const float* __restrict__ boxes,
                         const unsigned char* __restrict__ valid, const int* __restrict__ sval,
                         float* __restrict__ selS, float* __restrict__ selB,
                         float* __restrict__ nkey, int* __restrict__ nval) {
  int idx = blockIdx.x * 256 + threadIdx.x;
  if (idx >= 2 * NPADN) return;
  int b = idx / NPADN, sp = idx % NPADN;
  size_t nb = (size_t)b * NPADN + sp;
  if (sp >= SELN) { nkey[nb] = NEGINF; nval[nb] = sp; return; }
  const int selOff[5]  = {0, 1000, 2000, 3000, 4000};
  const int npadOff[5] = {0, 262144, 327680, 344064, 348160};
  const int aOff[5]    = {0, 182400, 228000, 239400, 242250};
  int lvl = sp < 1000 ? 0 : sp < 2000 ? 1 : sp < 3000 ? 2 : sp < 4000 ? 3 : 4;
  int rank = sp - selOff[lvl];
  int li = sval[(size_t)b * NPS + npadOff[lvl] + rank];
  size_t gi = (size_t)b * AT + aOff[lvl] + li;
  float sc = scores[gi];
  selS[(size_t)b * SELN + sp] = sc;
  float* sb = selB + ((size_t)b * SELN + sp) * 4;
  const float* bx = boxes + gi * 4;
  sb[0] = bx[0]; sb[1] = bx[1]; sb[2] = bx[2]; sb[3] = bx[3];
  nkey[nb] = valid[gi] ? sc : NEGINF;
  nval[nb] = sp;
}

// ---------------- IoU suppression bitmask over score-sorted boxes ----------------
__global__ void __launch_bounds__(160) k_iou(
    const float* __restrict__ nkey, const int* __restrict__ nval,
    const float* __restrict__ selB, unsigned int* __restrict__ mask) {
  int i = blockIdx.x, b = blockIdx.y, t = threadIdx.x;
  if (t >= MASKW) return;
  unsigned int bits = 0;
  float ki = nkey[(size_t)b * NPADN + i];
  if (ki > -1e38f && t < 149) {
    int si = nval[(size_t)b * NPADN + i];
    int lv = si < 1000 ? 0 : si < 2000 ? 1 : si < 3000 ? 2 : si < 4000 ? 3 : 4;
    float offi = 1217.0f * (float)lv;
    const float* bi = selB + ((size_t)b * SELN + si) * 4;
    float ax0 = bi[0] + offi, ay0 = bi[1] + offi, ax1 = bi[2] + offi, ay1 = bi[3] + offi;
    float areaA = (ax1 - ax0) * (ay1 - ay0);
    for (int u = 0; u < 32; ++u) {
      int j = t * 32 + u;
      if (j <= i || j >= SELN) continue;
      int sj = nval[(size_t)b * NPADN + j];
      int lv2 = sj < 1000 ? 0 : sj < 2000 ? 1 : sj < 3000 ? 2 : sj < 4000 ? 3 : 4;
      float offj = 1217.0f * (float)lv2;
      const float* bj = selB + ((size_t)b * SELN + sj) * 4;
      float bx0 = bj[0] + offj, by0 = bj[1] + offj, bx1 = bj[2] + offj, by1 = bj[3] + offj;
      float areaB = (bx1 - bx0) * (by1 - by0);
      float ix = fmaxf(fminf(ax1, bx1) - fmaxf(ax0, bx0), 0.0f);
      float iy = fmaxf(fminf(ay1, by1) - fmaxf(ay0, by0), 0.0f);
      float inter = ix * iy;
      float iou = inter / (areaA + areaB - inter + 1e-9f);
      if (iou > 0.7f) bits |= (1u << u);
    }
  }
  mask[((size_t)b * SELN + i) * MASKW + t] = bits;
}

// ---------------- greedy sequential NMS scan (matches reference fori_loop) ----------------
__global__ void __launch_bounds__(160) k_scan(
    const float* __restrict__ nkey, const unsigned int* __restrict__ mask,
    unsigned char* __restrict__ keep) {
  int b = blockIdx.x, t = threadIdx.x;
  __shared__ unsigned int remv[MASKW];
  __shared__ int flag;
  if (t < MASKW) remv[t] = 0u;
  __syncthreads();
  for (int i = 0; i < SELN; ++i) {
    if (t == 0) {
      bool sup = (remv[i >> 5] >> (i & 31)) & 1u;
      float ki = nkey[(size_t)b * NPADN + i];
      flag = (!sup && ki > -1e38f) ? 1 : 0;
      keep[(size_t)b * NPADN + i] = (unsigned char)flag;
    }
    __syncthreads();
    if (flag && t < 149) remv[t] |= mask[((size_t)b * SELN + i) * MASKW + t];
    __syncthreads();
  }
}

// ---------------- emit top POST_NMS kept boxes/scores ----------------
__global__ void k_finalize(const unsigned char* __restrict__ keep, const int* __restrict__ nval,
                           const float* __restrict__ selS, const float* __restrict__ selB,
                           float* __restrict__ out) {
  int b = blockIdx.x;
  if (threadIdx.x != 0) return;
  float* ob = out + (size_t)b * 4000;
  float* os = out + 8000 + (size_t)b * 1000;
  int cnt = 0;
  for (int i = 0; i < SELN && cnt < 1000; ++i) {
    if (keep[(size_t)b * NPADN + i]) {
      int sp = nval[(size_t)b * NPADN + i];
      const float* sb = selB + ((size_t)b * SELN + sp) * 4;
      ob[cnt * 4 + 0] = sb[0]; ob[cnt * 4 + 1] = sb[1];
      ob[cnt * 4 + 2] = sb[2]; ob[cnt * 4 + 3] = sb[3];
      os[cnt] = selS[(size_t)b * SELN + sp];
      ++cnt;
    }
  }
  for (; cnt < 1000; ++cnt) {
    ob[cnt * 4 + 0] = 0.0f; ob[cnt * 4 + 1] = 0.0f;
    ob[cnt * 4 + 2] = 0.0f; ob[cnt * 4 + 3] = 0.0f;
    os[cnt] = -1e9f;
  }
}

extern "C" void kernel_launch(void* const* d_in, const int* in_sizes, int n_in,
                              void* d_out, int out_size, void* d_ws, size_t ws_size,
                              hipStream_t stream) {
  (void)in_sizes; (void)n_in; (void)out_size; (void)ws_size;
  const float* feats[5];
  for (int i = 0; i < 5; ++i) feats[i] = (const float*)d_in[i];
  const float* conv_w = (const float*)d_in[5];
  const float* conv_b = (const float*)d_in[6];
  const float* cls_w  = (const float*)d_in[7];
  const float* cls_b  = (const float*)d_in[8];
  const float* box_w  = (const float*)d_in[9];
  const float* box_b  = (const float*)d_in[10];

  static const int   Hs[5]       = {200, 100, 50, 25, 13};
  static const int   Wd[5]       = {304, 152, 76, 38, 19};
  static const int   HWs[5]      = {60800, 15200, 3800, 950, 247};
  static const int   PoffT[5]    = {0, 60800, 76000, 79800, 80750};
  static const int   AoffT[5]    = {0, 182400, 228000, 239400, 242250};
  static const int   NpadT[5]    = {262144, 65536, 16384, 4096, 1024};
  static const int   NpadOffT[5] = {0, 262144, 327680, 344064, 348160};
  static const float strideT[5]  = {4.0f, 8.0f, 16.0f, 32.0f, 64.0f};
  static const float sizeT[5]    = {32.0f, 64.0f, 128.0f, 256.0f, 512.0f};

  char* ws = (char*)d_ws;
  size_t off = 0;
  auto alloc = [&](size_t bytes) { size_t o = off; off = (off + bytes + 255) & ~(size_t)255; return o; };
  unsigned short* fbf   = (unsigned short*)(ws + alloc((size_t)2 * PT * 256 * 2));
  unsigned short* tbf   = (unsigned short*)(ws + alloc((size_t)2 * PT * 256 * 2));
  unsigned short* wfrag = (unsigned short*)(ws + alloc((size_t)16 * 72 * 32 * 16 * 2));
  float*          scores = (float*)(ws + alloc((size_t)2 * AT * 4));
  float*          boxes  = (float*)(ws + alloc((size_t)2 * AT * 16));
  unsigned char*  valid  = (unsigned char*)(ws + alloc((size_t)2 * AT));
  float*          skey   = (float*)(ws + alloc((size_t)2 * NPS * 4));
  int*            sval   = (int*)(ws + alloc((size_t)2 * NPS * 4));
  float*          selS   = (float*)(ws + alloc((size_t)2 * SELN * 4));
  float*          selB   = (float*)(ws + alloc((size_t)2 * SELN * 16));
  float*          nkey   = (float*)(ws + alloc((size_t)2 * NPADN * 4));
  int*            nval   = (int*)(ws + alloc((size_t)2 * NPADN * 4));
  unsigned int*   mask   = (unsigned int*)(ws + alloc((size_t)2 * SELN * MASKW * 4));
  unsigned char*  keep   = (unsigned char*)(ws + alloc((size_t)2 * NPADN));

  {
    int total = 16 * 72 * 32 * 16;
    k_convert_weights<<<(total + 255) / 256, 256, 0, stream>>>(conv_w, wfrag);
  }
  for (int L = 0; L < 5; ++L) {
    int total = 2 * 32 * HWs[L];
    k_convert_input<<<(total + 255) / 256, 256, 0, stream>>>(feats[L], fbf, HWs[L], PoffT[L]);
  }
  for (int L = 0; L < 5; ++L) {
    dim3 g((HWs[L] + 63) / 64, 2, 2);
    k_conv3x3<<<g, 256, 0, stream>>>(fbf, wfrag, conv_b, tbf, Hs[L], Wd[L], PoffT[L]);
  }
  for (int L = 0; L < 5; ++L) {
    int total = 2 * HWs[L];
    k_heads<<<(total + 255) / 256, 256, 0, stream>>>(tbf, cls_w, cls_b, box_w, box_b,
        scores, boxes, valid, Hs[L], Wd[L], PoffT[L], AoffT[L], strideT[L], sizeT[L]);
  }
  for (int L = 0; L < 5; ++L) {
    int npad = NpadT[L];
    int total = 2 * npad;
    k_sortinit<<<(total + 255) / 256, 256, 0, stream>>>(scores, skey, sval,
        3 * HWs[L], AoffT[L], npad, NpadOffT[L]);
  }
  for (int L = 0; L < 5; ++L) {
    int n = NpadT[L];
    float* kk = skey + NpadOffT[L];
    int*   vv = sval + NpadOffT[L];
    for (int k = 2; k <= n; k <<= 1)
      for (int j = k >> 1; j > 0; j >>= 1)
        k_bitonic<<<((2 * n) + 255) / 256, 256, 0, stream>>>(kk, vv, NPS, n, k, j);
  }
  k_gather<<<(2 * NPADN + 255) / 256, 256, 0, stream>>>(scores, boxes, valid, sval,
                                                        selS, selB, nkey, nval);
  for (int k = 2; k <= NPADN; k <<= 1)
    for (int j = k >> 1; j > 0; j >>= 1)
      k_bitonic<<<((2 * NPADN) + 255) / 256, 256, 0, stream>>>(nkey, nval, NPADN, NPADN, k, j);
  k_iou<<<dim3(SELN, 2), 160, 0, stream>>>(nkey, nval, selB, mask);
  k_scan<<<2, 160, 0, stream>>>(nkey, mask, keep);
  k_finalize<<<2, 32, 0, stream>>>(keep, nval, selS, selB, (float*)d_out);
}
